// GroupedQueryAttention_75814762709711
// MI455X (gfx1250) — matrix operand order repaired and checked by reading, not compiled, not run on hardware
//
#include <hip/hip_runtime.h>
#include <hip/hip_bf16.h>

// ---------------------------------------------------------------------------
// GQA block for MI455X (gfx1250): bf16 WMMA GEMMs + flash attention,
// async global->LDS staging (ASYNCcnt) for contiguous tiles.
// B=2, S=2048, D=2048, H=32, G=8, DH=64, HG=4, kv_dim=512
// ---------------------------------------------------------------------------

typedef __attribute__((ext_vector_type(16))) __bf16 v16bf;
typedef __attribute__((ext_vector_type(8)))  float  v8f;

#define B_   2
#define S_   2048
#define D_   2048
#define G_   8
#define HG_  4
#define DH_  64
#define KV_  512
#define M_   (B_ * S_)   // 4096 rows for all projection GEMMs

// float -> bf16, round to nearest even (bit trick), stored as raw ushort
__device__ __forceinline__ unsigned short f2bf(float x) {
    unsigned int u = __float_as_uint(x);
    u += 0x7FFFu + ((u >> 16) & 1u);
    return (unsigned short)(u >> 16);
}

union FragU {
    v16bf v;
    uint4 q[2];
    unsigned short u[16];
};

// Async DMA: global (per-lane 64-bit address) -> LDS (per-lane 32-bit addr),
// 16 bytes per lane, tracked by ASYNCcnt (ISA 08_async_tensor §4).
__device__ __forceinline__ void async_copy_b128(const void* gptr, void* lptr) {
    unsigned lds = (unsigned)(size_t)lptr;                 // low 32b = LDS addr
    unsigned long long ga = (unsigned long long)(size_t)gptr;
    asm volatile("global_load_async_to_lds_b128 %0, %1, off"
                 :: "v"(lds), "v"(ga) : "memory");
}
__device__ __forceinline__ void async_wait_all() {
    asm volatile("s_wait_asynccnt 0" ::: "memory");
}

// A-fragment (16x32 bf16, MxK) from LDS row-major tile (lda elems; lda*2 must
// be 16B multiple). ISA 7.12.2: lane m=lane&15; K base=(lane>>4)*8;
// VGPR0..3 -> K=base..base+7, VGPR4..7 -> K=base+16..base+23.
__device__ __forceinline__ v16bf frag_a_from_lds(const unsigned short* p0,
                                                 int lda, int lane) {
    int m    = lane & 15;
    int base = (lane >> 4) * 8;
    const unsigned short* p = p0 + m * lda + base;
    FragU f;
    f.q[0] = *(const uint4*)(p);
    f.q[1] = *(const uint4*)(p + 16);
    return f.v;
}

// B-fragment (32x16 bf16, KxN) from LDS K-major tile: column n = n0 + (lane&15);
// K base=(lane>>4)*8; elems 0..7 -> K=base..base+7, 8..15 -> K=base+16..base+23.
__device__ __forceinline__ v16bf frag_b_from_lds(const unsigned short* p0,
                                                 int ldb, int lane, int n0) {
    const unsigned short* p = p0 + ((lane >> 4) * 8) * ldb + n0 + (lane & 15);
    FragU f;
#pragma unroll
    for (int s = 0; s < 8; ++s) {
        f.u[s]     = p[s * ldb];
        f.u[8 + s] = p[(16 + s) * ldb];
    }
    return f.v;
}

// ---------------------------------------------------------------------------
// Kernel 1: f32 -> bf16 cast (vectorized x4)
// ---------------------------------------------------------------------------
__global__ void cast_f32_to_bf16(const float* __restrict__ src,
                                 unsigned short* __restrict__ dst, int n4) {
    int i = blockIdx.x * blockDim.x + threadIdx.x;
    if (i >= n4) return;
    float4 f = ((const float4*)src)[i];
    ushort4 o;
    o.x = f2bf(f.x); o.y = f2bf(f.y); o.z = f2bf(f.z); o.w = f2bf(f.w);
    ((ushort4*)dst)[i] = o;
}

// ---------------------------------------------------------------------------
// Kernel 2: C[m,n] = sum_k A[m,k] * W[n,k]   (x @ W^T), bf16 in, fp32 acc.
// Block = 256 threads (8 waves, 4 M-groups x 2 N-groups).
// Block tile M=128, N=128, K-slab 64. Wave tile 32x64:
//   2 A-frags x 4 B-frags -> 8 accumulators, 16 WMMAs per K-slab.
// Every B-fragment feeds 2 WMMAs, every A-fragment feeds 4 (LDS reuse).
// ---------------------------------------------------------------------------
template <bool STORE_F32>
__global__ __launch_bounds__(256, 1)
void gemm_bf16_wmma(const unsigned short* __restrict__ A, // [M,K]
                    const unsigned short* __restrict__ W, // [N,K]
                    unsigned short* __restrict__ Cb,      // bf16 out
                    float* __restrict__ Cf,               // f32 out
                    int N, int K) {
    __shared__ unsigned short sA [128 * 72];   // [m][k] row-major (144B pitch)
    __shared__ unsigned short sBt[64 * 136];   // [k][n] K-major (272B pitch)

    const int tid  = threadIdx.x;
    const int lane = tid & 31;
    const int wave = tid >> 5;     // 0..7
    const int mw   = wave >> 1;    // 0..3 : M-group (32 rows each)
    const int nw   = wave & 1;     // 0..1 : N-group (64 cols each)
    const int m0   = blockIdx.y * 128;
    const int n0   = blockIdx.x * 128;

    v8f acc[2][4];
#pragma unroll
    for (int af = 0; af < 2; ++af)
#pragma unroll
        for (int j = 0; j < 4; ++j)
#pragma unroll
            for (int e = 0; e < 8; ++e) acc[af][j][e] = 0.0f;

    for (int k0 = 0; k0 < K; k0 += 64) {
        if (k0 + 64 < K) {  // speculative prefetch of next W slab
            __builtin_prefetch(W + (size_t)(n0 + (tid >> 1)) * K + k0 + 64, 0, 0);
        }
        __syncthreads();
        // Stage 128x64 A (async DMA) and 128x64 W (transposed via VGPRs).
        for (int c = tid; c < 1024; c += 256) {
            int r  = c >> 3;           // 0..127
            int cc = (c & 7) * 8;      // 0..56
            async_copy_b128(A + (size_t)(m0 + r) * K + k0 + cc,
                            sA + r * 72 + cc);
            uint4 wv = *(const uint4*)(W + (size_t)(n0 + r) * K + k0 + cc);
            const unsigned short* u = (const unsigned short*)&wv;
#pragma unroll
            for (int e = 0; e < 8; ++e) sBt[(cc + e) * 136 + r] = u[e];
        }
        async_wait_all();
        __syncthreads();

#pragma unroll
        for (int kk = 0; kk < 2; ++kk) {
            v16bf a0 = frag_a_from_lds(sA + (mw * 32 +  0) * 72 + kk * 32, 72, lane);
            v16bf a1 = frag_a_from_lds(sA + (mw * 32 + 16) * 72 + kk * 32, 72, lane);
#pragma unroll
            for (int j = 0; j < 4; ++j) {
                v16bf b = frag_b_from_lds(sBt + kk * 32 * 136, 136, lane,
                                          nw * 64 + j * 16);
                acc[0][j] = __builtin_amdgcn_wmma_f32_16x16x32_bf16(
                    false, a0, false, b, (short)0, acc[0][j], false, false);
                acc[1][j] = __builtin_amdgcn_wmma_f32_16x16x32_bf16(
                    false, a1, false, b, (short)0, acc[1][j], false, false);
            }
        }
    }

    // C layout: VGPR r, lanes 0-15 -> row r, lanes 16-31 -> row r+8; col=lane&15
    const int half = lane >> 4;
    const int col  = lane & 15;
#pragma unroll
    for (int af = 0; af < 2; ++af)
#pragma unroll
        for (int j = 0; j < 4; ++j)
#pragma unroll
            for (int r = 0; r < 8; ++r) {
                int row = m0 + mw * 32 + af * 16 + r + 8 * half;
                int cc  = n0 + nw * 64 + j * 16 + col;
                float vv = acc[af][j][r];
                if (STORE_F32) Cf[(size_t)row * N + cc] = vv;
                else           Cb[(size_t)row * N + cc] = f2bf(vv);
            }
}

// ---------------------------------------------------------------------------
// Kernel 3: flash attention for one (b, g, h) and one 64-row query tile.
// grid = (S/64, B*G*HG), block = 128 (4 waves, 16 query rows per wave).
// QP: [B*S, D] bf16 (head-major within D), KP/VP: [B*S, KV] bf16.
// ---------------------------------------------------------------------------
__global__ __launch_bounds__(128, 1)
void gqa_flash_wmma(const unsigned short* __restrict__ QP,
                    const unsigned short* __restrict__ KP,
                    const unsigned short* __restrict__ VP,
                    unsigned short* __restrict__ AO) {
    __shared__ unsigned short sQ [64 * 72];      // [s][d] row-major
    __shared__ unsigned short sKt[64 * 72];      // [d][t] K-major for QK^T
    __shared__ unsigned short sV [64 * 72];      // [t][d] K-major for PV
    __shared__ unsigned short sP [4][16 * 72];   // per-wave P staging

    const int tid  = threadIdx.x;
    const int lane = tid & 31;
    const int wave = tid >> 5;
    const int s0   = blockIdx.x * 64;
    const int y    = blockIdx.y;          // 0..63
    const int b    = y >> 5;              // 32 (g,h) pairs per batch
    const int g    = (y >> 2) & 7;
    const int h    = y & 3;
    const int qoff = (g * HG_ + h) * DH_;
    const int koff = g * DH_;
    const int half = lane >> 4;
    const int col  = lane & 15;
    const float scale = 0.125f;           // 1/sqrt(64)

    // Stage the 64x64 Q tile once via async DMA; fragments stay in registers.
    for (int c = tid; c < 512; c += 128) {
        int r = c >> 3, cc = (c & 7) * 8;
        async_copy_b128(QP + (size_t)(b * S_ + s0 + r) * D_ + qoff + cc,
                        sQ + r * 72 + cc);
    }
    async_wait_all();
    __syncthreads();
    v16bf aq0 = frag_a_from_lds(sQ + wave * 16 * 72, 72, lane);       // d 0..31
    v16bf aq1 = frag_a_from_lds(sQ + wave * 16 * 72 + 32, 72, lane);  // d 32..63

    float mrow[8], lrow[8];
    v8f o[4];
#pragma unroll
    for (int r = 0; r < 8; ++r) { mrow[r] = -1e30f; lrow[r] = 0.0f; }
#pragma unroll
    for (int jd = 0; jd < 4; ++jd)
#pragma unroll
        for (int e = 0; e < 8; ++e) o[jd][e] = 0.0f;

    for (int t0 = 0; t0 <= s0; t0 += 64) {
        __syncthreads();
        for (int c = tid; c < 512; c += 128) {
            int r = c >> 3, cc = (c & 7) * 8;
            // K tile transposed -> sKt[d][t] (through VGPRs)
            uint4 kv = *(const uint4*)(KP + (size_t)(b * S_ + t0 + r) * KV_ + koff + cc);
            const unsigned short* u = (const unsigned short*)&kv;
#pragma unroll
            for (int e = 0; e < 8; ++e) sKt[(cc + e) * 72 + r] = u[e];
            // V tile contiguous -> async DMA
            async_copy_b128(VP + (size_t)(b * S_ + t0 + r) * KV_ + koff + cc,
                            sV + r * 72 + cc);
        }
        async_wait_all();
        __syncthreads();

        // scores = Q K^T  (2 WMMAs per 16x16 subtile: d=0..31, d=32..63)
        v8f sc[4];
#pragma unroll
        for (int j = 0; j < 4; ++j) {
#pragma unroll
            for (int e = 0; e < 8; ++e) sc[j][e] = 0.0f;
            v16bf bk0 = frag_b_from_lds(sKt, 72, lane, j * 16);
            sc[j] = __builtin_amdgcn_wmma_f32_16x16x32_bf16(
                false, aq0, false, bk0, (short)0, sc[j], false, false);
            v16bf bk1 = frag_b_from_lds(sKt + 32 * 72, 72, lane, j * 16);
            sc[j] = __builtin_amdgcn_wmma_f32_16x16x32_bf16(
                false, aq1, false, bk1, (short)0, sc[j], false, false);
        }

        // scale + causal mask (only the diagonal tile needs masking)
        const bool diag = (t0 == s0);
#pragma unroll
        for (int j = 0; j < 4; ++j)
#pragma unroll
            for (int r = 0; r < 8; ++r) {
                float vv = sc[j][r] * scale;
                if (diag) {
                    int tg = t0 + j * 16 + col;
                    int sg = s0 + wave * 16 + r + 8 * half;
                    if (tg > sg) vv = -1e30f;
                }
                sc[j][r] = vv;
            }

        // online softmax per row (rows live across 16-lane halves)
#pragma unroll
        for (int r = 0; r < 8; ++r) {
            float mx = fmaxf(fmaxf(sc[0][r], sc[1][r]), fmaxf(sc[2][r], sc[3][r]));
#pragma unroll
            for (int off = 8; off; off >>= 1) mx = fmaxf(mx, __shfl_xor(mx, off, 16));
            float mnew  = fmaxf(mrow[r], mx);
            float alpha = __expf(mrow[r] - mnew);
            float rs = 0.0f;
            int prow = r + 8 * half;
#pragma unroll
            for (int j = 0; j < 4; ++j) {
                float p = __expf(sc[j][r] - mnew);
                rs += p;
                sP[wave][prow * 72 + j * 16 + col] = f2bf(p);
            }
#pragma unroll
            for (int off = 8; off; off >>= 1) rs += __shfl_xor(rs, off, 16);
            lrow[r] = lrow[r] * alpha + rs;
            mrow[r] = mnew;
#pragma unroll
            for (int jd = 0; jd < 4; ++jd) o[jd][r] *= alpha;
        }

        // PV: P (16x64) from per-wave LDS strip (same-wave LDS is in-order)
        v16bf ap0 = frag_a_from_lds(sP[wave], 72, lane);        // t 0..31
        v16bf ap1 = frag_a_from_lds(sP[wave] + 32, 72, lane);   // t 32..63
#pragma unroll
        for (int jd = 0; jd < 4; ++jd) {
            v16bf bv0 = frag_b_from_lds(sV, 72, lane, jd * 16);
            o[jd] = __builtin_amdgcn_wmma_f32_16x16x32_bf16(
                false, ap0, false, bv0, (short)0, o[jd], false, false);
            v16bf bv1 = frag_b_from_lds(sV + 32 * 72, 72, lane, jd * 16);
            o[jd] = __builtin_amdgcn_wmma_f32_16x16x32_bf16(
                false, ap1, false, bv1, (short)0, o[jd], false, false);
        }
    }

    // normalize and store bf16 attention output (head-major within D)
#pragma unroll
    for (int jd = 0; jd < 4; ++jd)
#pragma unroll
        for (int r = 0; r < 8; ++r) {
            int sg = s0 + wave * 16 + r + 8 * half;
            int d  = jd * 16 + col;
            AO[(size_t)(b * S_ + sg) * D_ + qoff + d] = f2bf(o[jd][r] / lrow[r]);
        }
}

// ---------------------------------------------------------------------------
// Host launcher
// ---------------------------------------------------------------------------
extern "C" void kernel_launch(void* const* d_in, const int* in_sizes, int n_in,
                              void* d_out, int out_size, void* d_ws, size_t ws_size,
                              hipStream_t stream) {
    (void)in_sizes; (void)n_in; (void)out_size; (void)ws_size;
    const float* q  = (const float*)d_in[0];
    const float* k  = (const float*)d_in[1];
    const float* v  = (const float*)d_in[2];
    /* d_in[3] = causal mask: handled analytically */
    const float* wq = (const float*)d_in[4];
    const float* wk = (const float*)d_in[5];
    const float* wv = (const float*)d_in[6];
    const float* wo = (const float*)d_in[7];

    unsigned short* ws = (unsigned short*)d_ws;
    size_t off = 0;
    auto alloc = [&](size_t n) { unsigned short* p = ws + off; off += n; return p; };
    const size_t MD = (size_t)M_ * D_;    // 8.39M elems
    const size_t DD = (size_t)D_ * D_;    // 4.19M
    const size_t KD = (size_t)KV_ * D_;   // 1.05M
    const size_t MK = (size_t)M_ * KV_;   // 2.10M

    unsigned short* qb  = alloc(MD);
    unsigned short* kb  = alloc(MD);
    unsigned short* vb  = alloc(MD);
    unsigned short* wqb = alloc(DD);
    unsigned short* wkb = alloc(KD);
    unsigned short* wvb = alloc(KD);
    unsigned short* wob = alloc(DD);
    unsigned short* qp  = alloc(MD);
    unsigned short* kp  = alloc(MK);
    unsigned short* vp  = alloc(MK);
    unsigned short* ao  = qb;  // qb is dead after the Q projection GEMM

    auto cast = [&](const float* s, unsigned short* d, size_t n) {
        int n4 = (int)(n / 4);
        cast_f32_to_bf16<<<(n4 + 255) / 256, 256, 0, stream>>>(s, d, n4);
    };
    cast(q,  qb,  MD);
    cast(k,  kb,  MD);
    cast(v,  vb,  MD);
    cast(wq, wqb, DD);
    cast(wk, wkb, KD);
    cast(wv, wvb, KD);
    cast(wo, wob, DD);

    dim3 blk(256);
    // projections: qp = q @ wq^T, kp = k @ wk^T, vp = v @ wv^T
    gemm_bf16_wmma<false><<<dim3(D_ / 128,  M_ / 128), blk, 0, stream>>>(
        qb, wqb, qp, nullptr, D_, D_);
    gemm_bf16_wmma<false><<<dim3(KV_ / 128, M_ / 128), blk, 0, stream>>>(
        kb, wkb, kp, nullptr, KV_, D_);
    gemm_bf16_wmma<false><<<dim3(KV_ / 128, M_ / 128), blk, 0, stream>>>(
        vb, wvb, vp, nullptr, KV_, D_);
    // grouped causal flash attention
    gqa_flash_wmma<<<dim3(S_ / 64, B_ * G_ * HG_), dim3(128), 0, stream>>>(
        qp, kp, vp, ao);
    // out projection -> fp32 result
    gemm_bf16_wmma<true><<<dim3(D_ / 128, M_ / 128), blk, 0, stream>>>(
        ao, wob, nullptr, (float*)d_out, M_, D_);
}
